// MaskedAttention_21552145891549
// MI455X (gfx1250) — matrix operand order repaired and checked by reading, not compiled, not run on hardware
//
#include <hip/hip_runtime.h>
#include <cmath>

typedef __bf16 bf16_t;
typedef __attribute__((ext_vector_type(16))) __bf16 bf16x16;
typedef __attribute__((ext_vector_type(8)))  __bf16 bf16x8;
typedef __attribute__((ext_vector_type(4)))  __bf16 bf16x4;
typedef __attribute__((ext_vector_type(8)))  float  f32x8;
typedef __attribute__((ext_vector_type(4)))  float  f32x4;

// generic-vector (gcc-style) int4 to match the async-LDS builtin prototype
typedef int i32x4_gv __attribute__((vector_size(16)));
typedef __attribute__((address_space(1))) i32x4_gv gvec4_t;   // printed as "v4i __device__*"
typedef __attribute__((address_space(3))) i32x4_gv lvec4_t;   // printed as "v4i __shared__*"

namespace {
constexpr int kB  = 4;
constexpr int kS  = 2048;
constexpr int kD  = 1024;
constexpr int kMS = kB * kS;              // 8192
constexpr int BM  = 256, BN = 128, BK = 32;
constexpr int BKP = BK + 8;               // 80B row stride: 16B-aligned, conflict-free b128
}

union FragU { bf16x16 v; bf16x8 h[2]; };

// ---- gfx1250 async global->LDS path (ASYNCcnt), with safe fallback ----------
#if __has_builtin(__builtin_amdgcn_global_load_async_to_lds_b128)
#define HAVE_ASYNC_LDS 1
#else
#define HAVE_ASYNC_LDS 0
#endif

__device__ __forceinline__ void async_copy_b128(const bf16_t* g, bf16_t* l) {
#if HAVE_ASYNC_LDS
  __builtin_amdgcn_global_load_async_to_lds_b128((gvec4_t*)g, (lvec4_t*)l, 0, 0);
#else
  *(bf16x8*)l = *(const bf16x8*)g;
#endif
}

__device__ __forceinline__ void wait_async_keep6() {
#if HAVE_ASYNC_LDS
#if __has_builtin(__builtin_amdgcn_s_wait_asynccnt)
  __builtin_amdgcn_s_wait_asynccnt(6);     // leave next stage (6 ops/wave) in flight
#else
  asm volatile("s_wait_asynccnt 0x6" ::: "memory");
#endif
#endif
}
__device__ __forceinline__ void wait_async_all() {
#if HAVE_ASYNC_LDS
#if __has_builtin(__builtin_amdgcn_s_wait_asynccnt)
  __builtin_amdgcn_s_wait_asynccnt(0);
#else
  asm volatile("s_wait_asynccnt 0x0" ::: "memory");
#endif
#endif
}

// ---------------------------------------------------------------------------
// Elementwise f32 -> bf16 (vectorized by 4)
// ---------------------------------------------------------------------------
__global__ __launch_bounds__(256)
void convert_f32_bf16_kernel(const float* __restrict__ src, bf16_t* __restrict__ dst) {
  const size_t i = ((size_t)blockIdx.x * 256 + threadIdx.x) * 4;
  const f32x4 v = *(const f32x4*)(src + i);
  bf16x4 o;
#pragma unroll
  for (int r = 0; r < 4; ++r) o[r] = (bf16_t)v[r];
  *(bf16x4*)(dst + i) = o;
}

// ---------------------------------------------------------------------------
// Transpose-convert W[k,n] (f32, DxD row-major) -> Wt[n,k] (bf16, row-major)
// ---------------------------------------------------------------------------
__global__ __launch_bounds__(256)
void transpose_convert_kernel(const float* __restrict__ W, bf16_t* __restrict__ Wt, int D) {
  __shared__ float tile[32][33];
  const int nBase = blockIdx.x * 32;
  const int kBase = blockIdx.y * 32;
  const int tx = threadIdx.x & 31;
  const int ty = threadIdx.x >> 5;
#pragma unroll
  for (int i = 0; i < 4; ++i)
    tile[ty + i * 8][tx] = W[(size_t)(kBase + ty + i * 8) * D + nBase + tx];
  __syncthreads();
#pragma unroll
  for (int i = 0; i < 4; ++i)
    Wt[(size_t)(nBase + ty + i * 8) * D + kBase + tx] = (bf16_t)tile[tx][ty + i * 8];
}

// ---------------------------------------------------------------------------
// NT GEMM: C[m,n] = scale * sum_k A[m,k]*Bt[n,k] (+ bias[n]);  A,Bt bf16.
// Block tile 256x128x32, 256 threads = 8 waves (4x2), wave tile 64x64 =
// 4x4 v_wmma_f32_16x16x32_bf16 per K-step. Double-buffered LDS filled by
// GLOBAL_LOAD_ASYNC_TO_LDS_B128 (6 ops/wave/stage), s_wait_asynccnt pipeline.
// ---------------------------------------------------------------------------
template <typename TOut, bool TRANS_OUT, bool CAUSAL_SKIP, bool CAUSAL_TRIMK>
__global__ __launch_bounds__(256)
void gemm_bf16_nt_kernel(const bf16_t* __restrict__ A, int lda,
                         const bf16_t* __restrict__ Bt, int ldb,
                         const float* __restrict__ bias,
                         TOut* __restrict__ C, int ldc,
                         int K, float scale) {
  __shared__ bf16_t As[2][BM * BKP];   // 2 x 20KB
  __shared__ bf16_t Bs[2][BN * BKP];   // 2 x 10KB

  const int mBase = blockIdx.y * BM;
  const int nBase = blockIdx.x * BN;
  if (CAUSAL_SKIP && nBase > mBase + (BM - 1)) return;   // fully-masked tile

  const int tid   = threadIdx.x;
  const int lane  = tid & 31;
  const int wave  = tid >> 5;
  const int m0    = (wave >> 1) * 64;  // 0,64,128,192
  const int n0    = (wave & 1) * 64;   // 0,64
  const int l15   = lane & 15;
  const int lhalf = lane >> 4;
  const int koffA = lhalf * 8;         // A frag K: {off..off+7, off+16..off+23}
  const int koffB = lhalf * 8;         // B frag K: {off..off+7, off+16..off+23}

  f32x8 acc[4][4];
#pragma unroll
  for (int i = 0; i < 4; ++i)
#pragma unroll
    for (int j = 0; j < 4; ++j)
#pragma unroll
      for (int r = 0; r < 8; ++r) acc[i][j][r] = 0.0f;

  const int kLimit = CAUSAL_TRIMK ? ((mBase + BM < K) ? (mBase + BM) : K) : K;
  const int numK   = kLimit / BK;

  // issue one LDS stage (A: 4 b128/thread, B: 2 b128/thread -> 6 async/wave)
  auto stage = [&](int buf, int kb) {
    const bf16_t* aSrc = A + (size_t)mBase * lda + kb;
    const bf16_t* bSrc = Bt + (size_t)nBase * ldb + kb;
#pragma unroll
    for (int it = 0; it < 4; ++it) {
      const int linear = tid + it * 256;            // 1024 b128 chunks
      const int row = linear >> 2;
      const int col = (linear & 3) * 8;
      async_copy_b128(aSrc + (size_t)row * lda + col, &As[buf][row * BKP + col]);
    }
#pragma unroll
    for (int it = 0; it < 2; ++it) {
      const int linear = tid + it * 256;            // 512 b128 chunks
      const int row = linear >> 2;
      const int col = (linear & 3) * 8;
      async_copy_b128(bSrc + (size_t)row * ldb + col, &Bs[buf][row * BKP + col]);
    }
  };

  stage(0, 0);
  for (int ki = 0; ki < numK; ++ki) {
    const int cur = ki & 1;
    if (ki + 1 < numK) {
      stage(cur ^ 1, (ki + 1) * BK);   // prefetch next tile into other buffer
      wait_async_keep6();              // current stage complete; next in flight
    } else {
      wait_async_all();
    }
    __syncthreads();

    const bf16_t* AsC = As[cur];
    const bf16_t* BsC = Bs[cur];
    bf16x16 bfrag[4];
#pragma unroll
    for (int j = 0; j < 4; ++j) {
      const int n = n0 + j * 16 + l15;
      FragU u;
      u.h[0] = *(const bf16x8*)&BsC[n * BKP + koffB];
      u.h[1] = *(const bf16x8*)&BsC[n * BKP + koffB + 16];
      bfrag[j] = u.v;
    }
#pragma unroll
    for (int i = 0; i < 4; ++i) {
      const int m = m0 + i * 16 + l15;
      FragU u;
      u.h[0] = *(const bf16x8*)&AsC[m * BKP + koffA];
      u.h[1] = *(const bf16x8*)&AsC[m * BKP + koffA + 16];
      const bf16x16 afrag = u.v;
#pragma unroll
      for (int j = 0; j < 4; ++j) {
        acc[i][j] = __builtin_amdgcn_wmma_f32_16x16x32_bf16(
            false, afrag, false, bfrag[j], (short)0, acc[i][j], false, false);
      }
    }
    __syncthreads();
  }

  // C/D layout: VGPR r <-> M = r + 8*lhalf, N = l15
#pragma unroll
  for (int i = 0; i < 4; ++i) {
#pragma unroll
    for (int j = 0; j < 4; ++j) {
      const int gmBase = mBase + m0 + i * 16 + lhalf * 8;
      const int gn     = nBase + n0 + j * 16 + l15;
      const float bv   = bias ? bias[gn] : 0.0f;
#pragma unroll
      for (int r = 0; r < 8; ++r) {
        const float v = acc[i][j][r] * scale + bv;
        const size_t idx =
            TRANS_OUT ? ((size_t)gn * ldc + (gmBase + r)) : ((size_t)(gmBase + r) * ldc + gn);
        C[idx] = (TOut)v;
      }
    }
  }
}

// ---------------------------------------------------------------------------
// Causal softmax: block per row s; logits f32 [S,S] -> probs bf16 [S,S]
// ---------------------------------------------------------------------------
__global__ __launch_bounds__(256)
void softmax_causal_kernel(const float* __restrict__ logits, bf16_t* __restrict__ probs, int S) {
  __shared__ float red[256];
  const int s   = blockIdx.x;
  const int tid = threadIdx.x;
  const float* row = logits + (size_t)s * S;

  float vals[8];
  float mx = -INFINITY;
#pragma unroll
  for (int e = 0; e < 8; ++e) {
    const int t = e * 256 + tid;
    const float x = (t <= s) ? row[t] : -INFINITY;
    vals[e] = x;
    mx = fmaxf(mx, x);
  }
  red[tid] = mx;
  __syncthreads();
  for (int off = 128; off > 0; off >>= 1) {
    if (tid < off) red[tid] = fmaxf(red[tid], red[tid + off]);
    __syncthreads();
  }
  mx = red[0];
  __syncthreads();

  float sum = 0.0f;
#pragma unroll
  for (int e = 0; e < 8; ++e) {
    const float ex = (vals[e] == -INFINITY) ? 0.0f : __expf(vals[e] - mx);
    vals[e] = ex;
    sum += ex;
  }
  red[tid] = sum;
  __syncthreads();
  for (int off = 128; off > 0; off >>= 1) {
    if (tid < off) red[tid] += red[tid + off];
    __syncthreads();
  }
  const float inv = 1.0f / red[0];

  bf16_t* prow = probs + (size_t)s * S;
#pragma unroll
  for (int e = 0; e < 8; ++e) {
    const int t = e * 256 + tid;
    prow[t] = (bf16_t)(vals[e] * inv);
  }
}

// ---------------------------------------------------------------------------
extern "C" void kernel_launch(void* const* d_in, const int* in_sizes, int n_in,
                              void* d_out, int out_size, void* d_ws, size_t ws_size,
                              hipStream_t stream) {
  (void)in_sizes; (void)n_in; (void)out_size; (void)ws_size;
  const float* q  = (const float*)d_in[0];
  const float* k  = (const float*)d_in[1];
  const float* v  = (const float*)d_in[2];
  // d_in[3] = mask — causality applied analytically
  const float* wq = (const float*)d_in[4];
  const float* bq = (const float*)d_in[5];
  const float* wk = (const float*)d_in[6];
  const float* bk = (const float*)d_in[7];
  const float* wv = (const float*)d_in[8];
  const float* bv = (const float*)d_in[9];
  const float* wo = (const float*)d_in[10];
  const float* bo = (const float*)d_in[11];
  float* out = (float*)d_out;

  char* ws = (char*)d_ws;
  size_t off = 0;
  auto alloc = [&](size_t bytes) -> char* {
    char* p = ws + off;
    off = (off + bytes + 255) & ~(size_t)255;
    return p;
  };
  bf16_t* WqT    = (bf16_t*)alloc((size_t)kD * kD * 2);
  bf16_t* WkT    = (bf16_t*)alloc((size_t)kD * kD * 2);
  bf16_t* WvT    = (bf16_t*)alloc((size_t)kD * kD * 2);
  bf16_t* WoT    = (bf16_t*)alloc((size_t)kD * kD * 2);
  bf16_t* qb     = (bf16_t*)alloc((size_t)kMS * kD * 2);   // bf16 copies of inputs
  bf16_t* kb     = (bf16_t*)alloc((size_t)kMS * kD * 2);
  bf16_t* vb     = (bf16_t*)alloc((size_t)kMS * kD * 2);
  bf16_t* Qb     = (bf16_t*)alloc((size_t)kMS * kD * 2);   // [B*S, D]
  bf16_t* Kb     = (bf16_t*)alloc((size_t)kMS * kD * 2);   // [B*S, D]
  bf16_t* Vt     = (bf16_t*)alloc((size_t)kD * kMS * 2);   // [D, B*S]
  bf16_t* ctx    = (bf16_t*)alloc((size_t)kMS * kD * 2);   // [B*S, D]
  float*  logits = (float*) alloc((size_t)kS * kS * 4);    // per-batch, reused
  bf16_t* probs  = (bf16_t*)alloc((size_t)kS * kS * 2);    // per-batch, reused

  const dim3 blk(256);

  const dim3 gCvt((kMS * kD / 4) / 256);
  convert_f32_bf16_kernel<<<gCvt, blk, 0, stream>>>(q, qb);
  convert_f32_bf16_kernel<<<gCvt, blk, 0, stream>>>(k, kb);
  convert_f32_bf16_kernel<<<gCvt, blk, 0, stream>>>(v, vb);

  const dim3 gT(kD / 32, kD / 32);
  transpose_convert_kernel<<<gT, blk, 0, stream>>>(wq, WqT, kD);
  transpose_convert_kernel<<<gT, blk, 0, stream>>>(wk, WkT, kD);
  transpose_convert_kernel<<<gT, blk, 0, stream>>>(wv, WvT, kD);
  transpose_convert_kernel<<<gT, blk, 0, stream>>>(wo, WoT, kD);

  const float invScale = 1.0f / sqrtf((float)kD);   // 1/32

  // Projections (V stored transposed for the NT P@V GEMM)
  const dim3 gProj(kD / BN, kMS / BM);
  gemm_bf16_nt_kernel<bf16_t, false, false, false><<<gProj, blk, 0, stream>>>(
      qb, kD, WqT, kD, bq, Qb, kD, kD, 1.0f);
  gemm_bf16_nt_kernel<bf16_t, false, false, false><<<gProj, blk, 0, stream>>>(
      kb, kD, WkT, kD, bk, Kb, kD, kD, 1.0f);
  gemm_bf16_nt_kernel<bf16_t, true, false, false><<<gProj, blk, 0, stream>>>(
      vb, kD, WvT, kD, bv, Vt, kMS, kD, 1.0f);

  // Attention, per batch (logits/probs buffers reused, stream-ordered)
  const dim3 gLogit(kS / BN, kS / BM);
  const dim3 gCtx(kD / BN, kS / BM);
  for (int b = 0; b < kB; ++b) {
    gemm_bf16_nt_kernel<float, false, true, false><<<gLogit, blk, 0, stream>>>(
        Qb + (size_t)b * kS * kD, kD, Kb + (size_t)b * kS * kD, kD, nullptr,
        logits, kS, kD, invScale);
    softmax_causal_kernel<<<dim3(kS), blk, 0, stream>>>(logits, probs, kS);
    gemm_bf16_nt_kernel<bf16_t, false, false, true><<<gCtx, blk, 0, stream>>>(
        probs, kS, Vt + (size_t)b * kS, kMS, nullptr,
        ctx + (size_t)b * kS * kD, kD, kS, 1.0f);
  }

  // Output projection
  gemm_bf16_nt_kernel<float, false, false, false><<<gProj, blk, 0, stream>>>(
      ctx, kD, WoT, kD, bo, out, kD, kD, 1.0f);
}
